// GPT_23459111371206
// MI455X (gfx1250) — matrix operand order repaired and checked by reading, not compiled, not run on hardware
//
#include <hip/hip_runtime.h>
#include <hip/hip_bf16.h>
#include <math.h>

// ---- model dims ----
#define Bb 2
#define Ss 1024
#define Dd 768
#define Hh 12
#define Ll 12
#define Vv 50257
#define HDd 64
#define Nn (Bb * Ss)

typedef __bf16 bf16;
typedef __attribute__((ext_vector_type(16))) __bf16 v16bf;
typedef __attribute__((ext_vector_type(8)))  float  v8f;

union Frag16 { uint4 u4[2]; v16bf v; };      // 32B: 16 bf16
union Pack8  { uint4 u4;    bf16  e[8]; };   // 16B: 8 bf16

// ---------------------------------------------------------------------------
// Embedding: x[b,s,:] = tok_emb[idx[b,s],:] + pos_emb[s,:]   (fp32 residual)
// ---------------------------------------------------------------------------
__global__ void embed_kernel(const int* __restrict__ idx,
                             const float* __restrict__ tok,
                             const float* __restrict__ pos,
                             float* __restrict__ x) {
    int bs = blockIdx.x;            // 0..N-1
    int s  = bs % Ss;
    int t  = idx[bs];
    const float* te = tok + (size_t)t * Dd;
    const float* pe = pos + (size_t)s * Dd;
    float* xo = x + (size_t)bs * Dd;
    for (int d = threadIdx.x; d < Dd; d += blockDim.x)
        xo[d] = te[d] + pe[d];
}

// ---------------------------------------------------------------------------
// LayerNorm (fp32 math) -> bf16 output (single conversion point)
// ---------------------------------------------------------------------------
__global__ void ln_kernel(const float* __restrict__ x,
                          const float* __restrict__ sc,
                          const float* __restrict__ bi,
                          bf16* __restrict__ out) {
    int n = blockIdx.x;
    const float* xr = x + (size_t)n * Dd;
    float s = 0.f, s2 = 0.f;
    for (int d = threadIdx.x; d < Dd; d += blockDim.x) {
        float v = xr[d];
        s += v; s2 += v * v;
    }
    __shared__ float rs[8], rs2[8];
    for (int o = 16; o > 0; o >>= 1) {
        s  += __shfl_down(s,  o, 32);
        s2 += __shfl_down(s2, o, 32);
    }
    int wid = threadIdx.x >> 5;
    if ((threadIdx.x & 31) == 0) { rs[wid] = s; rs2[wid] = s2; }
    __syncthreads();
    if (threadIdx.x == 0) {
        float a = 0.f, b = 0.f;
        for (int i = 0; i < 8; i++) { a += rs[i]; b += rs2[i]; }
        rs[0] = a; rs2[0] = b;
    }
    __syncthreads();
    float mean = rs[0] * (1.0f / Dd);
    float var  = rs2[0] * (1.0f / Dd) - mean * mean;
    float inv  = rsqrtf(var + 1e-5f);
    bf16* o = out + (size_t)n * Dd;
    for (int d = threadIdx.x; d < Dd; d += blockDim.x)
        o[d] = (bf16)((xr[d] - mean) * inv * sc[d] + bi[d]);
}

__device__ __forceinline__ float gelu_exact(float v) {
    return 0.5f * v * (1.0f + erff(v * 0.70710678118654752f));
}

// ---------------------------------------------------------------------------
// GEMM via WMMA bf16 (fp32 accumulate), LDS-staged weights:
//   C[n,m] = act( sum_k A[n,k] * W[m,k] + bias[m] ) (+ resid)
// A: [N,K] bf16 row-major (pre-converted). W: [M,K] fp32 row-major.
// Block = 256 threads = 8 waves. Block tile: 128 rows x 64 cols.
// Per k-step (K32): 256 threads cooperatively convert the 64x32 fp32 weight
// tile to bf16 in LDS (4 cvt_pk per thread); each wave loads one A fragment
// (2x b128, no conversion) and reuses it across 4 WMMAs whose B fragments
// come from LDS (ds_load_b128, conflict-free 80B row stride).
// Out-of-range weight rows are clamped (their columns are never stored).
// Output: fp32 (+bias/act/resid) or bf16 (+bias/act), per pointers.
// grid = (ceil(M/64), N/128).
// ---------------------------------------------------------------------------
__global__ void gemm_bf16_wmma(const bf16* __restrict__ A,
                               const float* __restrict__ W,
                               const float* __restrict__ bias,
                               const float* __restrict__ resid,
                               float* __restrict__ Cf,
                               bf16* __restrict__ Cb,
                               int K, int M, int act) {
    int tid  = threadIdx.x;
    int wv   = tid >> 5;        // wave 0..7 -> row tile
    int lane = tid & 31;
    int hf   = lane >> 4;
    int lrow = lane & 15;
    int m0   = blockIdx.x * 64;
    int row0 = blockIdx.y * 128;

    // staging role: thread -> (weight row r64, k-part kp)
    int r64 = tid >> 2;                 // 0..63
    int kp  = (tid & 3) * 8;            // 0,8,16,24
    int wr  = m0 + r64;
    if (wr > M - 1) wr = M - 1;
    const float* wsrc = W + (size_t)wr * K + kp;

    // A fragment base: row (row0 + wv*16 + lrow)
    const bf16* ap = A + (size_t)(row0 + wv * 16 + lrow) * K;

    __shared__ __align__(16) bf16 bwt[64][40];   // 64 rows x 32 k, stride 40 (80B)

    v8f acc[4] = {};
    for (int k0 = 0; k0 < K; k0 += 32) {
        // ---- cooperative stage: fp32 weights -> bf16 LDS tile
        {
            const float4* w4 = (const float4*)(wsrc + k0);
            float4 w0 = w4[0], w1 = w4[1];
            Pack8 pk;
            pk.e[0] = (bf16)w0.x; pk.e[1] = (bf16)w0.y;
            pk.e[2] = (bf16)w0.z; pk.e[3] = (bf16)w0.w;
            pk.e[4] = (bf16)w1.x; pk.e[5] = (bf16)w1.y;
            pk.e[6] = (bf16)w1.z; pk.e[7] = (bf16)w1.w;
            *(uint4*)&bwt[r64][kp] = pk.u4;
        }
        __syncthreads();

        // ---- A fragment: 16 bf16, direct b128 loads (A-frag lane layout)
        Frag16 af;
        af.u4[0] = *(const uint4*)(ap + k0 + hf * 8);
        af.u4[1] = *(const uint4*)(ap + k0 + 16 + hf * 8);

        // ---- 4 column tiles: B frags from LDS + WMMA
#pragma unroll
        for (int t = 0; t < 4; t++) {
            Frag16 bw;
            const bf16* bp = &bwt[t * 16 + lrow][hf * 8];
            bw.u4[0] = *(const uint4*)bp;
            bw.u4[1] = *(const uint4*)(bp + 16);
            acc[t] = __builtin_amdgcn_wmma_f32_16x16x32_bf16(
                false, af.v, false, bw.v, (short)0, acc[t], false, false);
        }
        __syncthreads();
    }

    // ---- epilogue
    int tnrow = row0 + wv * 16;
#pragma unroll
    for (int t = 0; t < 4; t++) {
        int col = m0 + t * 16 + lrow;            // same column for both halves
        if (col < M) {
            float bv = bias ? bias[col] : 0.0f;
#pragma unroll
            for (int r = 0; r < 8; r++) {
                size_t row = (size_t)(tnrow + r + hf * 8);
                float v = acc[t][r] + bv;
                if (act == 1) v = gelu_exact(v);
                if (Cf) {
                    if (resid) v += resid[row * (size_t)M + col];
                    Cf[row * (size_t)M + col] = v;
                } else {
                    Cb[row * (size_t)M + col] = (bf16)v;
                }
            }
        }
    }
}

// ---------------------------------------------------------------------------
// Flash-style causal attention, one wave per (query-16-tile, head, batch).
// qkv: [B, S, 3D] bf16 (q | k | v thirds). attn: [B, S, D] bf16.
// Fragments are direct bf16 copies (no cvt); 1/sqrt(HD) applied to fp32
// scores; softmax fp32 online; P transposed C->A layout through LDS.
// ---------------------------------------------------------------------------
__global__ void attn_kernel(const bf16* __restrict__ qkv,
                            bf16* __restrict__ attn) {
    int lane = threadIdx.x;
    int hf   = lane >> 4;
    int lrow = lane & 15;
    int qt = blockIdx.x;        // S/16
    int h  = blockIdx.y;        // H
    int b  = blockIdx.z;        // B

    const bf16* base = qkv + (size_t)b * Ss * 3 * Dd;
    const bf16* qb = base + h * HDd;
    const bf16* kb = base + Dd + h * HDd;
    const bf16* vb = base + 2 * Dd + h * HDd;

    // Q A-frags for hd-chunks [0,32) and [32,64): direct b128 loads
    const bf16* qp = qb + (size_t)(qt * 16 + lrow) * 3 * Dd;
    Frag16 aq0, aq1;
    aq0.u4[0] = *(const uint4*)(qp + hf * 16);
    aq0.u4[1] = *(const uint4*)(qp + hf * 16 + 8);
    aq1.u4[0] = *(const uint4*)(qp + 32 + hf * 16);
    aq1.u4[1] = *(const uint4*)(qp + 32 + hf * 16 + 8);

    float m[8], l[8];
    v8f acc[4] = {};
#pragma unroll
    for (int r = 0; r < 8; r++) { m[r] = -1e30f; l[r] = 0.0f; }

    __shared__ float pls[16][34];   // 16x32 P tile (+pad), C-layout -> A-layout

    int qmax = qt * 16 + 15;
    for (int j0 = 0; j0 <= qmax; j0 += 32) {
        // ---- scores for 32 keys: two 16-col tiles, hd=64 split in 2 WMMAs
        v8f c0 = {}, c1 = {};
        {
            const bf16* kp0 = kb + (size_t)(j0 + lrow) * 3 * Dd;
            const bf16* kp1 = kb + (size_t)(j0 + 16 + lrow) * 3 * Dd;
            Frag16 b00, b01, b10, b11;
            b00.u4[0] = *(const uint4*)(kp0 + hf * 16);
            b00.u4[1] = *(const uint4*)(kp0 + hf * 16 + 8);
            b01.u4[0] = *(const uint4*)(kp0 + 32 + hf * 16);
            b01.u4[1] = *(const uint4*)(kp0 + 32 + hf * 16 + 8);
            b10.u4[0] = *(const uint4*)(kp1 + hf * 16);
            b10.u4[1] = *(const uint4*)(kp1 + hf * 16 + 8);
            b11.u4[0] = *(const uint4*)(kp1 + 32 + hf * 16);
            b11.u4[1] = *(const uint4*)(kp1 + 32 + hf * 16 + 8);
            c0 = __builtin_amdgcn_wmma_f32_16x16x32_bf16(false, aq0.v, false, b00.v, (short)0, c0, false, false);
            c0 = __builtin_amdgcn_wmma_f32_16x16x32_bf16(false, aq1.v, false, b01.v, (short)0, c0, false, false);
            c1 = __builtin_amdgcn_wmma_f32_16x16x32_bf16(false, aq0.v, false, b10.v, (short)0, c1, false, false);
            c1 = __builtin_amdgcn_wmma_f32_16x16x32_bf16(false, aq1.v, false, b11.v, (short)0, c1, false, false);
        }

        // ---- scale + causal mask + online softmax (fp32), write P to LDS
        int col0 = j0 + lrow;
        int col1 = j0 + 16 + lrow;
#pragma unroll
        for (int r = 0; r < 8; r++) {
            int qrow = qt * 16 + r + hf * 8;
            float s0 = (col0 <= qrow) ? c0[r] * 0.125f : -1e30f;
            float s1 = (col1 <= qrow) ? c1[r] * 0.125f : -1e30f;
            float mx = fmaxf(s0, s1);
#pragma unroll
            for (int o = 1; o < 16; o <<= 1)
                mx = fmaxf(mx, __shfl_xor(mx, o, 32));   // stays in 16-lane half
            float mnew = fmaxf(m[r], mx);
            float p0 = __expf(s0 - mnew);
            float p1 = __expf(s1 - mnew);
            float corr = __expf(m[r] - mnew);
            float rsum = p0 + p1;
#pragma unroll
            for (int o = 1; o < 16; o <<= 1)
                rsum += __shfl_xor(rsum, o, 32);
            l[r] = l[r] * corr + rsum;
            m[r] = mnew;
#pragma unroll
            for (int t = 0; t < 4; t++) acc[t][r] *= corr;
            pls[r + hf * 8][lrow]      = p0;
            pls[r + hf * 8][16 + lrow] = p1;
        }
        __syncthreads();

        // ---- re-read P in A-frag layout (16x32)
        v16bf pa;
#pragma unroll
        for (int i = 0; i < 8; i++) {
            pa[i]     = (bf16)pls[lrow][hf * 8 + i];
            pa[8 + i] = (bf16)pls[lrow][16 + hf * 8 + i];
        }
        __syncthreads();

        // ---- acc += P(16x32) * V(32x64) : 4 column tiles (bf16 copies)
#pragma unroll
        for (int t = 0; t < 4; t++) {
            v16bf bv;
#pragma unroll
            for (int e = 0; e < 16; e++) {
                int key = j0 + (e < 8 ? hf * 8 + e : 16 + hf * 8 + (e - 8));
                bv[e] = vb[(size_t)key * 3 * Dd + t * 16 + lrow];
            }
            acc[t] = __builtin_amdgcn_wmma_f32_16x16x32_bf16(
                false, pa, false, bv, (short)0, acc[t], false, false);
        }
    }

    // ---- normalize and store bf16 [B,S,D]
    bf16* ao = attn + (size_t)(b * Ss) * Dd + h * HDd;
#pragma unroll
    for (int r = 0; r < 8; r++) {
        int qrow = qt * 16 + r + hf * 8;
        float invl = 1.0f / l[r];
#pragma unroll
        for (int t = 0; t < 4; t++)
            ao[(size_t)qrow * Dd + t * 16 + lrow] = (bf16)(acc[t][r] * invl);
    }
}

// ---------------------------------------------------------------------------
extern "C" void kernel_launch(void* const* d_in, const int* in_sizes, int n_in,
                              void* d_out, int out_size, void* d_ws, size_t ws_size,
                              hipStream_t stream) {
    (void)in_sizes; (void)n_in; (void)out_size; (void)ws_size;

    const int*   idx   = (const int*)  d_in[0];
    const float* tok   = (const float*)d_in[1];
    const float* pos   = (const float*)d_in[2];
    const float* qkv_w = (const float*)d_in[3];
    const float* out_w = (const float*)d_in[4];
    const float* out_b = (const float*)d_in[5];
    const float* ln1_s = (const float*)d_in[6];
    const float* ln1_b = (const float*)d_in[7];
    const float* ln2_s = (const float*)d_in[8];
    const float* ln2_b = (const float*)d_in[9];
    const float* ff_w1 = (const float*)d_in[10];
    const float* ff_b1 = (const float*)d_in[11];
    const float* ff_w2 = (const float*)d_in[12];
    const float* ff_b2 = (const float*)d_in[13];
    const float* fn_s  = (const float*)d_in[14];
    const float* fn_b  = (const float*)d_in[15];
    float* out = (float*)d_out;

    // workspace carve (bytes): fp32 residual x + bf16 activation buffers
    char* wsb = (char*)d_ws;
    float* x     = (float*)wsb;                 wsb += (size_t)Nn * Dd * 4;
    bf16*  hbuf  = (bf16*)wsb;                  wsb += (size_t)Nn * Dd * 2;
    bf16*  qkvb  = (bf16*)wsb;                  wsb += (size_t)Nn * 3 * Dd * 2;
    bf16*  attnb = (bf16*)wsb;                  wsb += (size_t)Nn * Dd * 2;
    bf16*  ffhb  = (bf16*)wsb;

    embed_kernel<<<Nn, 256, 0, stream>>>(idx, tok, pos, x);

    for (int l = 0; l < Ll; l++) {
        // h = LN1(x) -> bf16 ; qkv = h @ qkv_w.T -> bf16
        ln_kernel<<<Nn, 256, 0, stream>>>(x, ln1_s + l * Dd, ln1_b + l * Dd, hbuf);
        gemm_bf16_wmma<<<dim3(3 * Dd / 64, Nn / 128), 256, 0, stream>>>(
            hbuf, qkv_w + (size_t)l * 3 * Dd * Dd, nullptr, nullptr,
            nullptr, qkvb, Dd, 3 * Dd, 0);

        // attention (bf16 in / bf16 out)
        attn_kernel<<<dim3(Ss / 16, Hh, Bb), 32, 0, stream>>>(qkvb, attnb);

        // x = x + attn @ out_w.T + out_b   (fp32 residual)
        gemm_bf16_wmma<<<dim3(Dd / 64, Nn / 128), 256, 0, stream>>>(
            attnb, out_w + (size_t)l * Dd * Dd, out_b + l * Dd, x,
            x, nullptr, Dd, Dd, 0);

        // h = LN2(x) -> bf16 ; ffh = gelu(h @ ff_w1.T + ff_b1) -> bf16
        ln_kernel<<<Nn, 256, 0, stream>>>(x, ln2_s + l * Dd, ln2_b + l * Dd, hbuf);
        gemm_bf16_wmma<<<dim3(4 * Dd / 64, Nn / 128), 256, 0, stream>>>(
            hbuf, ff_w1 + (size_t)l * 4 * Dd * Dd, ff_b1 + l * 4 * Dd, nullptr,
            nullptr, ffhb, Dd, 4 * Dd, 1);

        // x = x + ffh @ ff_w2.T + ff_b2    (fp32 residual)
        gemm_bf16_wmma<<<dim3(Dd / 64, Nn / 128), 256, 0, stream>>>(
            ffhb, ff_w2 + (size_t)l * Dd * 4 * Dd, ff_b2 + l * Dd, x,
            x, nullptr, 4 * Dd, Dd, 0);
    }

    // final LN + tied lm-head: out = LN(x) @ tok_emb.T -> fp32 [N, V]
    ln_kernel<<<Nn, 256, 0, stream>>>(x, fn_s, fn_b, hbuf);
    gemm_bf16_wmma<<<dim3((Vv + 63) / 64, Nn / 128), 256, 0, stream>>>(
        hbuf, tok, nullptr, nullptr, out, nullptr, Dd, Vv, 0);
}
